// GCN_60928406061379
// MI455X (gfx1250) — matrix operand order repaired and checked by reading, not compiled, not run on hardware
//
#include <hip/hip_runtime.h>

// GCN forward for MI455X (gfx1250): bf16 WMMA (f32 accumulate), wave32.
// out = log_softmax( adj @ relu( adj @ (feature@W1) + b1 ) @ W2 + b2 )
//
// All intermediates are stored TRANSPOSED (k-contiguous) and K-padded to 10016
// with exact zeros, so the streaming GEMM loops have no guards at all:
//  - A (adj) loads use clamped addresses (garbage rows/cols are harmless:
//    garbage rows -> store-guarded outputs; garbage k-cols multiply zero B rows)
//  - B tiles are contiguous b128 copies global->LDS (async-to-LDS if available)

#define NROW  10000
#define NROWP 10016   // K padded to multiple of 32 (pad rows are zeroed)
#define FIN   512
#define HID   256
#define NCLS  40
#define NCLSP 48      // 40 padded to 3x16
#define BM    64      // rows per workgroup
#define KT    32      // K tile (bf16 WMMA K)

typedef __attribute__((ext_vector_type(16))) __bf16 v16bf;
typedef __attribute__((ext_vector_type(8)))  __bf16 v8bf;
typedef __attribute__((ext_vector_type(8)))  float  v8f;
typedef __attribute__((ext_vector_type(4)))  float  v4f;
typedef __attribute__((ext_vector_type(4)))  int    v4i;

#define AS1 __attribute__((address_space(1)))
#define AS3 __attribute__((address_space(3)))

#if defined(__gfx1250__) && __has_builtin(__builtin_amdgcn_global_load_async_to_lds_b128)
#define USE_ASYNC 1
#else
#define USE_ASYNC 0
#endif

// 16-byte global->LDS copy (async-to-LDS when the builtin exists)
__device__ __forceinline__ void copy16_g2l(void* lds, const void* g) {
#if USE_ASYNC
  __builtin_amdgcn_global_load_async_to_lds_b128(
      (AS1 v4i*)const_cast<void*>(g), (AS3 v4i*)lds, 0, 0);
#else
  *(v8bf*)lds = *(const v8bf*)g;
#endif
}

__device__ __forceinline__ void async_wait() {
#if USE_ASYNC
#if __has_builtin(__builtin_amdgcn_s_wait_asynccnt)
  __builtin_amdgcn_s_wait_asynccnt(0);
#else
  asm volatile("s_wait_asynccnt 0x0" ::: "memory");
#endif
#endif
}

// ---- fragment loaders (ISA 7.12.2 layouts) ------------------------------
// A fragment 16x32 (MxK) from row-major bf16 tile (stride ld).
__device__ __forceinline__ v16bf load_a_frag(const __bf16* t, int ld, int m0,
                                             int kb, int lane) {
  const int m  = m0 + (lane & 15);
  const int kh = (lane >> 4) << 3;                 // 0 or 8
  const __bf16* p = t + m * ld + kb + kh;
  v8bf lo = *(const v8bf*)(p);
  v8bf hi = *(const v8bf*)(p + 16);
  v16bf f;
#pragma unroll
  for (int i = 0; i < 8; ++i) { f[i] = lo[i]; f[i + 8] = hi[i]; }
  return f;
}

// B fragment 32x16 (KxN) from k-contiguous ([n][k]) bf16 tile, row stride ld.
__device__ __forceinline__ v16bf load_b_frag(const __bf16* bt, int ld, int n0,
                                             int kb, int lane) {
  const int n  = n0 + (lane & 15);
  const int kh = (lane >> 4) << 3;                 // 0 or 8
  const __bf16* p = bt + n * ld + kb + kh;
  v8bf lo = *(const v8bf*)(p);
  v8bf hi = *(const v8bf*)(p + 16);
  v16bf f;
#pragma unroll
  for (int i = 0; i < 8; ++i) { f[i] = lo[i]; f[i + 8] = hi[i]; }
  return f;
}

// Stage a 64x32 f32 tile -> bf16 LDS (row-major, stride KT), branch-free.
// Addresses are clamped; callers guarantee garbage elements are harmless.
__device__ __forceinline__ void stage_a_f32(__bf16* As, const float* __restrict__ src,
                                            int row0, size_t ldg, int kcol0,
                                            int rowmax, int kmax8, int tid) {
  int r  = tid >> 2;
  int c0 = (tid & 3) << 3;
  int gr = row0 + r;   if (gr > rowmax) gr = rowmax;
  int gc = kcol0 + c0; if (gc > kmax8)  gc = kmax8;
  const float* gp = src + (size_t)gr * ldg + gc;
  v4f x0 = *(const v4f*)gp;
  v4f x1 = *(const v4f*)(gp + 4);
  v8bf o;
#pragma unroll
  for (int i = 0; i < 4; ++i) { o[i] = (__bf16)x0[i]; o[i + 4] = (__bf16)x1[i]; }
  *(v8bf*)&As[r * KT + c0] = o;
}

// ---- Kernel 0: W1t(bf16)[HID][FIN] = transpose(W1) ----------------------
__global__ __launch_bounds__(256) void gcn_k0(const float* __restrict__ W1,
                                              __bf16* __restrict__ W1t) {
  int i = blockIdx.x * 256 + threadIdx.x;
  if (i < HID * FIN) {
    int n = i / FIN, k = i - n * FIN;
    W1t[(size_t)n * FIN + k] = (__bf16)W1[(size_t)k * HID + n];
  }
}

// ---- Kernel 1: Ht(bf16)[HID][NROWP] = (feature @ W1)^T, pad rows zero ---
__global__ __launch_bounds__(256) void gcn_k1(const float* __restrict__ feat,
                                              const __bf16* __restrict__ W1t,
                                              __bf16* __restrict__ Ht) {
  __shared__ alignas(16) __bf16 As[BM * KT];    // 4 KB
  __shared__ alignas(16) __bf16 Bt[HID * KT];   // 16 KB ([n][k])
  const int tid  = threadIdx.x;
  const int lane = tid & 31;
  const int w    = tid >> 5;
  const int slab = w >> 1;                      // 0..3
  const int half = w & 1;                       // 0..1
  const int rb   = blockIdx.x * BM;

  v8f acc[8] = {};

  for (int kt = 0; kt < FIN; kt += KT) {
    stage_a_f32(As, feat, rb, FIN, kt, NROW - 1, FIN - 8, tid);
#pragma unroll
    for (int j = 0; j < 4; ++j) {               // B: 256x32 contiguous chunks
      int idx = j * 256 + tid;
      int n = idx >> 2, kk0 = (idx & 3) << 3;
      copy16_g2l(&Bt[n * KT + kk0], W1t + (size_t)n * FIN + kt + kk0);
    }
    async_wait();
    __syncthreads();
    v16bf a = load_a_frag(As, KT, slab * 16, 0, lane);
#pragma unroll
    for (int t = 0; t < 8; ++t) {
      v16bf b = load_b_frag(Bt, KT, half * 128 + t * 16, 0, lane);
      acc[t] = __builtin_amdgcn_wmma_f32_16x16x32_bf16(
          false, a, false, b, (short)0, acc[t], false, false);
    }
    __syncthreads();
  }

  const int col   = lane & 15;
  const int rbase = (lane >> 4) << 3;
#pragma unroll
  for (int t = 0; t < 8; ++t) {
    int n = half * 128 + t * 16 + col;
#pragma unroll
    for (int r = 0; r < 8; ++r) {
      int gr = rb + slab * 16 + rbase + r;
      if (gr < NROWP)
        Ht[(size_t)n * NROWP + gr] = (__bf16)((gr < NROW) ? acc[t][r] : 0.f);
    }
  }
}

// ---- Kernel 2: X1 = relu(adj@H + b1); H2t(bf16)[NCLSP][NROWP] = (X1@W2)^T
__global__ __launch_bounds__(256) void gcn_k2(const float* __restrict__ adj,
                                              const __bf16* __restrict__ Ht,
                                              const float* __restrict__ b1,
                                              const float* __restrict__ W2,
                                              __bf16* __restrict__ H2t) {
  __shared__ alignas(16) __bf16 As[BM * KT];       // 4 KB
  __shared__ alignas(16) __bf16 Bt[HID * KT];      // 16 KB
  __shared__ alignas(16) __bf16 X1s[BM * HID];     // 32 KB
  __shared__ alignas(16) __bf16 W2t[NCLSP * HID];  // 24 KB ([n][k])
  const int tid  = threadIdx.x;
  const int lane = tid & 31;
  const int w    = tid >> 5;
  const int slab = w >> 1;
  const int half = w & 1;
  const int rb   = blockIdx.x * BM;

  for (int i = tid; i < NCLSP * HID; i += 256) {   // stage W2 (pad cols 40..47)
    int n = i / HID, k = i - n * HID;
    W2t[n * HID + k] = (__bf16)((n < NCLS) ? W2[(size_t)k * NCLS + n] : 0.f);
  }

  v8f acc[8] = {};

  for (int kt = 0; kt < NROWP; kt += KT) {         // 313 iters, branch-free
    stage_a_f32(As, adj, rb, NROW, kt, NROW - 1, NROW - 8, tid);
#pragma unroll
    for (int j = 0; j < 4; ++j) {                  // B: Ht tile, contiguous
      int idx = j * 256 + tid;
      int n = idx >> 2, kk0 = (idx & 3) << 3;
      copy16_g2l(&Bt[n * KT + kk0], Ht + (size_t)n * NROWP + kt + kk0);
    }
    {
      int pr = rb + (tid >> 2);   if (pr > NROW - 1) pr = NROW - 1;
      int pk = kt + KT + ((tid & 3) << 3);
      if (pk <= NROW - 8) __builtin_prefetch(&adj[(size_t)pr * NROW + pk], 0, 0);
    }
    async_wait();
    __syncthreads();
    v16bf a = load_a_frag(As, KT, slab * 16, 0, lane);
#pragma unroll
    for (int t = 0; t < 8; ++t) {
      v16bf b = load_b_frag(Bt, KT, half * 128 + t * 16, 0, lane);
      acc[t] = __builtin_amdgcn_wmma_f32_16x16x32_bf16(
          false, a, false, b, (short)0, acc[t], false, false);
    }
    __syncthreads();
  }

  // epilogue: +b1, relu, stage X1 (bf16) to LDS
  const int col   = lane & 15;
  const int rbase = (lane >> 4) << 3;
#pragma unroll
  for (int t = 0; t < 8; ++t) {
    int n = half * 128 + t * 16 + col;
    float bias = b1[n];
#pragma unroll
    for (int r = 0; r < 8; ++r) {
      float v = acc[t][r] + bias;
      v = v > 0.f ? v : 0.f;
      X1s[(slab * 16 + rbase + r) * HID + n] = (__bf16)v;
    }
  }
  __syncthreads();

  // fused tiny GEMM: H2[64][48] = X1[64][256] @ W2pad[256][48], store ^T
  if (w < 6) {
#pragma unroll
    for (int tt = 0; tt < 2; ++tt) {
      int tile = w * 2 + tt;
      int slab2 = tile / 3, nt = tile - slab2 * 3;
      v8f c = {};
#pragma unroll
      for (int kb = 0; kb < HID; kb += KT) {
        v16bf a = load_a_frag(X1s, HID, slab2 * 16, kb, lane);
        v16bf b = load_b_frag(W2t, HID, nt * 16, kb, lane);
        c = __builtin_amdgcn_wmma_f32_16x16x32_bf16(
            false, a, false, b, (short)0, c, false, false);
      }
#pragma unroll
      for (int r = 0; r < 8; ++r) {
        int gr = rb + slab2 * 16 + rbase + r;
        if (gr < NROWP)
          H2t[(size_t)(nt * 16 + col) * NROWP + gr] =
              (__bf16)((gr < NROW) ? c[r] : 0.f);
      }
    }
  }
}

// ---- Kernel 3: OUT = log_softmax(adj @ H2 + b2) -------------------------
__global__ __launch_bounds__(256) void gcn_k3(const float* __restrict__ adj,
                                              const __bf16* __restrict__ H2t,
                                              const float* __restrict__ b2,
                                              float* __restrict__ out) {
  __shared__ alignas(16) __bf16 As[BM * KT];       // 4 KB
  __shared__ alignas(16) __bf16 Bt[NCLSP * KT];    // 3 KB
  __shared__ float Ls[BM * NCLSP];                 // 12 KB logits
  const int tid  = threadIdx.x;
  const int lane = tid & 31;
  const int w    = tid >> 5;
  const int slab = w >> 1;
  const int odd  = w & 1;                          // even wave: tiles 0,1; odd: 2
  const int rb   = blockIdx.x * BM;

  v8f accA = {}, accB = {};

  for (int kt = 0; kt < NROWP; kt += KT) {
    stage_a_f32(As, adj, rb, NROW, kt, NROW - 1, NROW - 8, tid);
    if (tid < 192) {                               // B: H2t tile, contiguous
      int n = tid >> 2, kk0 = (tid & 3) << 3;
      copy16_g2l(&Bt[n * KT + kk0], H2t + (size_t)n * NROWP + kt + kk0);
    }
    {
      int pr = rb + (tid >> 2);   if (pr > NROW - 1) pr = NROW - 1;
      int pk = kt + KT + ((tid & 3) << 3);
      if (pk <= NROW - 8) __builtin_prefetch(&adj[(size_t)pr * NROW + pk], 0, 0);
    }
    async_wait();
    __syncthreads();
    v16bf a = load_a_frag(As, KT, slab * 16, 0, lane);
    if (!odd) {
      v16bf bf0 = load_b_frag(Bt, KT, 0, 0, lane);
      accA = __builtin_amdgcn_wmma_f32_16x16x32_bf16(
          false, a, false, bf0, (short)0, accA, false, false);
      v16bf bf1 = load_b_frag(Bt, KT, 16, 0, lane);
      accB = __builtin_amdgcn_wmma_f32_16x16x32_bf16(
          false, a, false, bf1, (short)0, accB, false, false);
    } else {
      v16bf bf2 = load_b_frag(Bt, KT, 32, 0, lane);
      accA = __builtin_amdgcn_wmma_f32_16x16x32_bf16(
          false, a, false, bf2, (short)0, accA, false, false);
    }
    __syncthreads();
  }

  // logits (+b2) -> LDS
  const int col   = lane & 15;
  const int rbase = (lane >> 4) << 3;
  if (!odd) {
    float biasA = b2[col];
    float biasB = b2[col + 16];
#pragma unroll
    for (int r = 0; r < 8; ++r) {
      int row = slab * 16 + rbase + r;
      Ls[row * NCLSP + col]      = accA[r] + biasA;
      Ls[row * NCLSP + col + 16] = accB[r] + biasB;
    }
  } else {
    int n = col + 32;
    float bias = (n < NCLS) ? b2[n] : 0.f;
#pragma unroll
    for (int r = 0; r < 8; ++r) {
      int row = slab * 16 + rbase + r;
      Ls[row * NCLSP + n] = accA[r] + bias;
    }
  }
  __syncthreads();

  // per-row log_softmax over the real 40 classes
  if (tid < BM) {
    int gr = rb + tid;
    if (gr < NROW) {
      const float* lr = Ls + tid * NCLSP;
      float m = lr[0];
#pragma unroll
      for (int n = 1; n < NCLS; ++n) m = fmaxf(m, lr[n]);
      float s = 0.f;
#pragma unroll
      for (int n = 0; n < NCLS; ++n) s += __expf(lr[n] - m);
      float ls = m + __logf(s);
#pragma unroll
      for (int n = 0; n < NCLS; ++n)
        out[(size_t)gr * NCLS + n] = lr[n] - ls;
    }
  }
}

// ---- host launcher ------------------------------------------------------
extern "C" void kernel_launch(void* const* d_in, const int* in_sizes, int n_in,
                              void* d_out, int out_size, void* d_ws, size_t ws_size,
                              hipStream_t stream) {
  const float* feat = (const float*)d_in[0];   // [10000,512]
  const float* adj  = (const float*)d_in[1];   // [10000,10000]
  const float* W1   = (const float*)d_in[2];   // [512,256]
  const float* b1   = (const float*)d_in[3];   // [256]
  const float* W2   = (const float*)d_in[4];   // [256,40]
  const float* b2   = (const float*)d_in[5];   // [40]
  float* out = (float*)d_out;                  // [10000,40]

  char* ws = (char*)d_ws;
  __bf16* W1t = (__bf16*)ws;                                   // 0.25 MB
  __bf16* Ht  = (__bf16*)(ws + (size_t)HID * FIN * 2);         // 5.13 MB
  __bf16* H2t = (__bf16*)(ws + (size_t)HID * FIN * 2
                             + (size_t)HID * NROWP * 2);       // 0.96 MB

  const int grid = (NROW + BM - 1) / BM;       // 157
  gcn_k0<<<(HID * FIN + 255) / 256, 256, 0, stream>>>(W1, W1t);
  gcn_k1<<<grid, 256, 0, stream>>>(feat, W1t, Ht);
  gcn_k2<<<grid, 256, 0, stream>>>(adj, Ht, b1, W2, H2t);
  gcn_k3<<<grid, 256, 0, stream>>>(adj, H2t, b2, out);
}
